// QMLP_61993557950530
// MI455X (gfx1250) — matrix operand order repaired and checked by reading, not compiled, not run on hardware
//
#include <hip/hip_runtime.h>

// QMLP fused kernel for MI455X (gfx1250, wave32, WMMA).
// Transposed formulation Z_l = W_l x Z_{l-1}: W = A-matrix (fp16 hi+lo
// split; W0/W2 register-resident, W1 streamed from LDS), activations =
// B-matrix (exact fp16 hi+lo split of 12-bit fixed-point values).
// The inter-layer D->B conversion is shuffle-free: accumulator rows 8h+r
// of each 16-row tile pair already sit in B's table K order, matching the
// next layer's weight-fragment layout, so each lane builds its B operand
// purely from its own accumulator registers, in register-pair order that
// vectorizes into v_pk_add_f32 / v_cvt_pk_f16_f32 with no staging moves.

typedef __attribute__((ext_vector_type(16))) _Float16 v16h;
typedef __attribute__((ext_vector_type(8)))  float    v8f;
typedef __attribute__((ext_vector_type(4)))  float    v4f;

#define RND_C 49152.0f            // 1.5*2^15: (x+C)-C == RNE to 1/256 grid
#define HI_C  98304.0f            // 1.5*2^16: (q+C)-C == RNE to 1/128 grid
#define QMAXV (16.0f - 1.0f/256.0f)

// Round-to-nearest-even onto the 1/256 fixed-point grid (|x| < 16K).
__device__ __forceinline__ float qround(float x) {
  float t = x + RND_C;
  return t - RND_C;
}

// quantize = round + clamp[lob, qmax]; lob=0 fuses the ReLU of hidden layers.
__device__ __forceinline__ float quantc(float x, float lob) {
  return __builtin_amdgcn_fmed3f(qround(x), lob, QMAXV);
}

// Split grid value q (multiple of 1/256, |q| <= 16) into hi + lo where
// hi = RNE(q to 1/128 grid) and lo = q - hi in {0, +-1/256}. Both are
// exactly representable in fp16, and hi + lo == q exactly.
__device__ __forceinline__ void qsplit(float q, float& hi, float& lo) {
  float t = q + HI_C;
  hi = t - HI_C;
  lo = q - hi;
}

struct Frag { v16h hi; v16h lo; };

// D = A*B + C in fp16 with hi/lo error compensation (drop lo*lo term).
__device__ __forceinline__ v8f wmma3(const v16h& whi, const v16h& wlo,
                                     const v16h& bhi, const v16h& blo,
                                     v8f acc) {
  acc = __builtin_amdgcn_wmma_f32_16x16x32_f16(false, whi, false, bhi,
                                               (short)0, acc, false, false);
  acc = __builtin_amdgcn_wmma_f32_16x16x32_f16(false, wlo, false, bhi,
                                               (short)0, acc, false, false);
  acc = __builtin_amdgcn_wmma_f32_16x16x32_f16(false, whi, false, blo,
                                               (short)0, acc, false, false);
  return acc;
}

__device__ __forceinline__ void split16(const float* v, Frag& f) {
#pragma unroll
  for (int j = 0; j < 16; ++j) {
    _Float16 h = (_Float16)v[j];
    f.hi[j] = h;
    f.lo[j] = (_Float16)(v[j] - (float)h);
  }
}

// Standard A-layout fragment (for W0, whose B input has identity K order):
// lanes 0-15 row m=lane, K chunks at koff=0 / 16; lanes 16-31 row m=lane-16,
// K chunks at koff=8 / 24.
__device__ __forceinline__ void load_wfrag_std(const float* __restrict__ W,
                                               int ldk, int mtile, int kb,
                                               int lane, Frag& f) {
  const int row  = mtile * 16 + (lane & 15);
  const int koff = (lane & 16) ? 8 : 0;
  const float* p = W + (size_t)row * ldk + kb * 32 + koff;
  float v[16];
  *(v4f*)&v[0]  = *(const v4f*)(p);
  *(v4f*)&v[4]  = *(const v4f*)(p + 4);
  *(v4f*)&v[8]  = *(const v4f*)(p + 16);
  *(v4f*)&v[12] = *(const v4f*)(p + 20);
  split16(v, f);
}

// A-layout fragment, same as the standard one (for W1/W2, whose B input rows are the previous
// layer's D tiles in-place). Table K order: lanes 0-15 read cols
// kb*32+0..7,16..23 of row m=lane, lanes 16-31 read cols kb*32+8..15,24..31
// of row m=lane-16.
__device__ __forceinline__ void load_wfrag_perm(const float* __restrict__ W,
                                                int ldk, int mtile, int kb,
                                                int lane, Frag& f) {
  const int row = mtile * 16 + (lane & 15);
  const float* p =
      W + (size_t)row * ldk + kb * 32 + ((lane & 16) ? 8 : 0);
  float v[16];
  *(v4f*)&v[0]  = *(const v4f*)(p);
  *(v4f*)&v[4]  = *(const v4f*)(p + 4);
  *(v4f*)&v[8]  = *(const v4f*)(p + 16);
  *(v4f*)&v[12] = *(const v4f*)(p + 20);
  split16(v, f);
}

// Shuffle-free D->B conversion in B's table K order. Processing
// all of d0 (B elements 0..7) before all of d1 (B elements 8..15) keeps
// the SLP pairs inside one accumulator's consecutive registers, so the
// pk_add/cvt_pk vectorization needs no v_mov staging.
__device__ __forceinline__ void pack_b(const v8f& d0, const v8f& d1,
                                       float lob, v16h& bhi, v16h& blo) {
#pragma unroll
  for (int j = 0; j < 8; ++j) {
    float q = quantc(d0[j], lob);
    float h, l;
    qsplit(q, h, l);
    bhi[j] = (_Float16)h;
    blo[j] = (_Float16)l;
  }
#pragma unroll
  for (int j = 0; j < 8; ++j) {
    float q = quantc(d1[j], lob);
    float h, l;
    qsplit(q, h, l);
    bhi[8 + j] = (_Float16)h;
    blo[8 + j] = (_Float16)l;
  }
}

__global__ __launch_bounds__(256, 1)
void qmlp_fused_kernel(const float* __restrict__ inp,
                       const float* __restrict__ W0,
                       const float* __restrict__ W1,
                       const float* __restrict__ W2,
                       float* __restrict__ out, int ntiles) {
  const int  lane  = threadIdx.x & 31;
  const bool upper = (lane & 16) != 0;
  const int  wid = (int)((blockIdx.x * blockDim.x + threadIdx.x) >> 5);
  const int  nw  = (int)((gridDim.x * blockDim.x) >> 5);

  // W1 fragments live in LDS (pre-formatted per lane), streamed per use.
  // Index: [(t*2+kb)*2 + (0=hi,1=lo)][lane]. 16 KB total. Rolled setup
  // loop keeps the one-time formatting code small.
  __shared__ v16h w1sh[16][32];
  if (threadIdx.x < 32) {
#pragma unroll 1
    for (int f = 0; f < 8; ++f) {        // f = t*2 + kb
      Frag fr;
      load_wfrag_perm(W1, 64, f >> 1, f & 1, lane, fr);
      w1sh[f * 2 + 0][lane] = fr.hi;
      w1sh[f * 2 + 1][lane] = fr.lo;
    }
  }

  // Register-resident W0 / W2 fragments (loaded once per wave, L2-served).
  Frag w0f[4];   // 64x32: standard layout (identity K order from memory)
  Frag w2f[2];   // 16x64: standard layout (input = layer-1 D tiles)
#pragma unroll
  for (int t = 0; t < 4; ++t) load_wfrag_std(W0, 32, t, 0, lane, w0f[t]);
#pragma unroll
  for (int kb = 0; kb < 2; ++kb) load_wfrag_perm(W2, 64, 0, kb, lane, w2f[kb]);

  __syncthreads();

  for (int tile = wid; tile < ntiles; tile += nw) {
    // ---- load + quantize 16x32 input tile as B fragment (Z0 = X^T) ----
    const float* xp =
        inp + ((size_t)tile * 16 + (lane & 15)) * 32 + (upper ? 8 : 0);
    if (tile + nw < ntiles)
      __builtin_prefetch(inp + ((size_t)(tile + nw) * 16 + (lane & 15)) * 32,
                         0, 0);
    float x[16];
    *(v4f*)&x[0]  = *(const v4f*)(xp);
    *(v4f*)&x[4]  = *(const v4f*)(xp + 4);
    *(v4f*)&x[8]  = *(const v4f*)(xp + 16);
    *(v4f*)&x[12] = *(const v4f*)(xp + 20);
    v16h b0hi, b0lo;
#pragma unroll
    for (int j = 0; j < 16; ++j) {
      float q = quantc(x[j], -16.0f);         // exact 12-bit fixed point
      float h, l;
      qsplit(q, h, l);
      b0hi[j] = (_Float16)h;
      b0lo[j] = (_Float16)l;
    }

    // ---- layer 0: Z1 = W0 x Z0 (64x16, 12 WMMAs) ----
    v8f z1[4];
#pragma unroll
    for (int t = 0; t < 4; ++t) {
      v8f c = {};
      z1[t] = wmma3(w0f[t].hi, w0f[t].lo, b0hi, b0lo, c);
    }

    // ---- layer 1: Z2 = W1 x relu(q(Z1)) (64x16, 24 WMMAs, W1 via LDS) ----
    v16h b1hi[2], b1lo[2];
#pragma unroll
    for (int kb = 0; kb < 2; ++kb)
      pack_b(z1[2 * kb], z1[2 * kb + 1], 0.0f, b1hi[kb], b1lo[kb]);
    v8f z2[4];
#pragma unroll
    for (int t = 0; t < 4; ++t) {
      v8f c = {};
#pragma unroll
      for (int kb = 0; kb < 2; ++kb) {
        v16h whi = w1sh[(t * 2 + kb) * 2 + 0][lane];
        v16h wlo = w1sh[(t * 2 + kb) * 2 + 1][lane];
        c = wmma3(whi, wlo, b1hi[kb], b1lo[kb], c);
      }
      z2[t] = c;
      __builtin_amdgcn_sched_barrier(0);  // keep streamed frags near uses
    }

    // ---- layer 2: Z3 = W2 x relu(q(Z2)) (16x16, 6 WMMAs) ----
    v16h b2hi[2], b2lo[2];
#pragma unroll
    for (int kb = 0; kb < 2; ++kb)
      pack_b(z2[2 * kb], z2[2 * kb + 1], 0.0f, b2hi[kb], b2lo[kb]);
    v8f z3 = {};
#pragma unroll
    for (int kb = 0; kb < 2; ++kb)
      z3 = wmma3(w2f[kb].hi, w2f[kb].lo, b2hi[kb], b2lo[kb], z3);

    // ---- quantize + store Y = Z3^T: 16 rows x 16 features, coalesced ----
    float o[8];
#pragma unroll
    for (int r = 0; r < 8; ++r) o[r] = quantc(z3[r], -16.0f);
    float* op = out + ((size_t)tile * 16 + (lane & 15)) * 16 + (upper ? 8 : 0);
    *(v4f*)(op)     = *(const v4f*)&o[0];
    *(v4f*)(op + 4) = *(const v4f*)&o[4];
  }
}

extern "C" void kernel_launch(void* const* d_in, const int* in_sizes, int n_in,
                              void* d_out, int out_size, void* d_ws,
                              size_t ws_size, hipStream_t stream) {
  (void)n_in; (void)out_size; (void)d_ws; (void)ws_size;
  const float* inp = (const float*)d_in[0];
  const float* W0  = (const float*)d_in[1];
  const float* W1  = (const float*)d_in[2];
  const float* W2  = (const float*)d_in[3];
  float* out = (float*)d_out;

  const int B      = in_sizes[0] / 32;   // rows (D_IN = 32)
  const int ntiles = B / 16;             // 16-row tiles per wave iteration

  int blocks = (ntiles + 63) / 64;       // 8 waves/block, ~8 tiles/wave
  if (blocks > 1024) blocks = 1024;
  if (blocks < 1) blocks = 1;

  qmlp_fused_kernel<<<blocks, 256, 0, stream>>>(inp, W0, W1, W2, out, ntiles);
}
